// MyPointConv_45981919871335
// MI455X (gfx1250) — matrix operand order repaired and checked by reading, not compiled, not run on hardware
//
#include <hip/hip_runtime.h>
#include <hip/hip_bf16.h>

// ---------------------------------------------------------------------------
// MyPointConv on gfx1250 (MI455X): edge MLP via v_wmma_f32_16x16x32_f16,
// scatter-max via global_atomic_max_u32 on non-negative float bit patterns,
// node GEMM via WMMA.
// ---------------------------------------------------------------------------

typedef __attribute__((ext_vector_type(16))) _Float16 v16h;
typedef __attribute__((ext_vector_type(8)))  _Float16 v8h;
typedef __attribute__((ext_vector_type(4)))  _Float16 v4h;
typedef __attribute__((ext_vector_type(8)))  float    v8f;

#define N_NODES 50000
#define N_EDGES 1600000
#define EA      (N_EDGES + N_NODES)   // 1,650,000 edges incl. self loops

// workspace layout (bytes)
#define AGG_OFF   0u                  // 50000*128 f32 = 25,600,000 B
#define LW1P_OFF  25600000u           // 3*4*32*16 f16 = 12,288 B
#define LW2P_OFF  25612288u           // 2*8*32*16 f16 = 16,384 B
#define GW1P_OFF  25628672u           // 4*16*32*16 f16 = 65,536 B

// ---------------------------------------------------------------------------
// WMMA helpers
// ---------------------------------------------------------------------------
__device__ __forceinline__ v8f wmma_f16(v16h a, v16h b, v8f c) {
  // D(f32 16x16) = A(f16 16x32) * B(f16 32x16) + C
  return __builtin_amdgcn_wmma_f32_16x16x32_f16(
      /*neg_a=*/false, a, /*neg_b=*/false, b,
      /*c_mod=*/(short)0, c, /*reuse_a=*/false, /*reuse_b=*/false);
}

// A fragment from a row-major LDS tile [16][strideH] of f16.
// ISA 16-bit A layout: lane L holds row M=L&15; halves 0..7 -> K = 8*(L>>4)+0..7,
// halves 8..15 -> K = 16 + 8*(L>>4) + 0..7 (relative to kbase).
__device__ __forceinline__ v16h load_a_frag_lds(const _Float16* base, int strideH,
                                                int kbase, int lane) {
  const int row  = lane & 15;
  const int koff = (lane >> 4) << 3;            // 0 or 8
  const _Float16* p = base + row * strideH + kbase + koff;
  v8h lo = *(const v8h*)p;                      // halves 0..7  (16B aligned)
  v8h hi = *(const v8h*)(p + 16);               // halves 8..15 (16B aligned)
  v16h a;
#pragma unroll
  for (int i = 0; i < 8; ++i) { a[i] = lo[i]; a[i + 8] = hi[i]; }
  return a;
}

// B fragment from the pre-packed weight buffer: 16 contiguous halves per
// (kt,nt,lane) tuple -> single 32B load.
__device__ __forceinline__ v16h load_b_frag(const _Float16* __restrict__ packed,
                                            int tile, int lane) {
  return *(const v16h*)(packed + ((size_t)tile * 32 + lane) * 16);
}

// ---------------------------------------------------------------------------
// Prep kernels
// ---------------------------------------------------------------------------
__global__ void zero_agg_kernel(float* __restrict__ agg, int n) {
  int i = blockIdx.x * blockDim.x + threadIdx.x;
  if (i < n) agg[i] = 0.0f;
}

// Pack W[K][N] (f32, row-major) into B-fragment layout f16:
//   P[((kt*NT + nt)*32 + lane)*16 + j] = W[kt*32 + 8*(lane>>4) + (j&7) + 16*(j>>3)][nt*16 + (lane&15)]
// with zero padding for k >= K.
__global__ void pack_w_kernel(const float* __restrict__ W, _Float16* __restrict__ P,
                              int K, int N, int NT, int total) {
  int idx = blockIdx.x * blockDim.x + threadIdx.x;
  if (idx >= total) return;
  const int j    = idx & 15;
  const int lane = (idx >> 4) & 31;
  const int t    = idx >> 9;            // kt*NT + nt
  const int nt   = t % NT;
  const int kt   = t / NT;
  const int k = kt * 32 + ((lane >> 4) << 3) + (j & 7) + ((j >> 3) << 4);
  const int n = nt * 16 + (lane & 15);
  P[idx] = (k < K) ? (_Float16)W[(size_t)k * N + n] : (_Float16)0.0f;
}

// ---------------------------------------------------------------------------
// Edge kernel: 8 waves/block, 16 edges per wave.
// msg[16x96] (67 padded to 96) -> GEMM1 -> h1[16x64] -> GEMM2 -> scatter max.
// ---------------------------------------------------------------------------
__global__ __launch_bounds__(256)
void edge_mlp_scatter(const float* __restrict__ x, const float* __restrict__ pos,
                      const long long* __restrict__ src, const long long* __restrict__ dst,
                      const _Float16* __restrict__ lw1p, const float* __restrict__ lb1,
                      const _Float16* __restrict__ lw2p, const float* __restrict__ lb2,
                      unsigned int* __restrict__ agg) {
  __shared__ _Float16 smem[8 * 16 * 96];   // 24 KB: per-wave message / h1 tile
  __shared__ int sdst_s[8 * 16];

  const int tid  = threadIdx.x;
  const int wave = tid >> 5;
  const int lane = tid & 31;
  _Float16* smsg = smem + wave * (16 * 96);
  int* sd = sdst_s + wave * 16;
  const long long tileBase = (long long)(blockIdx.x * 8 + wave) * 16;

  // ---- build message tile: lane covers row (lane&15), features fb..fb+47 ----
  {
    const int r  = lane & 15;
    const int fb = (lane >> 4) * 48;
    const long long e = tileBase + r;
    float px = 0.f, py = 0.f, pz = 0.f;
    long long s = 0;
    int valid = 0;
    if (e < (long long)EA) {
      long long d;
      if (e < (long long)N_EDGES) { s = src[e]; d = dst[e]; }
      else                        { s = e - N_EDGES; d = s; }
      px = pos[s * 3 + 0] - pos[d * 3 + 0];
      py = pos[s * 3 + 1] - pos[d * 3 + 1];
      pz = pos[s * 3 + 2] - pos[d * 3 + 2];
      valid = 1;
      if (lane < 16) sd[r] = (int)d;
    } else if (lane < 16) {
      sd[r] = -1;
    }
    const float4* xr = (const float4*)(x + s * 64);
#pragma unroll
    for (int q = 0; q < 12; ++q) {
      const int f = fb + q * 4;
      float4 v = make_float4(0.f, 0.f, 0.f, 0.f);
      if (valid) {
        if (f < 64)       v = xr[f >> 2];          // node features (L2-resident gather)
        else if (f == 64) v = make_float4(px, py, pz, 0.f);  // pos diff, feat 67 = pad 0
      }
      v4h h;
      h[0] = (_Float16)v.x; h[1] = (_Float16)v.y;
      h[2] = (_Float16)v.z; h[3] = (_Float16)v.w;
      *(v4h*)(smsg + r * 96 + f) = h;              // 8B-aligned ds_store_b64
    }
  }
  __syncthreads();

  // ---- GEMM1: [16x96] @ [96x64] -> [16x64], K tiles of 32 ----
  v16h a1[3];
#pragma unroll
  for (int kt = 0; kt < 3; ++kt) a1[kt] = load_a_frag_lds(smsg, 96, kt * 32, lane);

  v8f c1[4];
#pragma unroll
  for (int nt = 0; nt < 4; ++nt) {
    v8f c = {};
#pragma unroll
    for (int kt = 0; kt < 3; ++kt)
      c = wmma_f16(a1[kt], load_b_frag(lw1p, kt * 4 + nt, lane), c);
    c1[nt] = c;
  }

  // bias + relu, stage h1 (f16, [16][64]) back into this wave's LDS region
  const int colBase = lane & 15;
  const int rowOff  = (lane >> 4) * 8;
#pragma unroll
  for (int nt = 0; nt < 4; ++nt) {
    const int col = nt * 16 + colBase;
    const float b = lb1[col];
#pragma unroll
    for (int r = 0; r < 8; ++r) {
      float v = c1[nt][r] + b;
      v = v > 0.f ? v : 0.f;
      smsg[(rowOff + r) * 64 + col] = (_Float16)v;
    }
  }
  __syncthreads();

  // ---- GEMM2: [16x64] @ [64x128] -> [16x128], then scatter-max ----
  v16h a2[2];
  a2[0] = load_a_frag_lds(smsg, 64, 0, lane);
  a2[1] = load_a_frag_lds(smsg, 64, 32, lane);

#pragma unroll
  for (int nt = 0; nt < 8; ++nt) {
    v8f c = {};
    c = wmma_f16(a2[0], load_b_frag(lw2p, 0 * 8 + nt, lane), c);
    c = wmma_f16(a2[1], load_b_frag(lw2p, 1 * 8 + nt, lane), c);
    const int col = nt * 16 + colBase;
    const float bb = lb2[col];
#pragma unroll
    for (int r = 0; r < 8; ++r) {
      float v = c[r] + bb;
      v = v > 0.f ? v : 0.f;           // relu => v >= 0 => uint bits order-preserving
      const int d = sd[rowOff + r];
      if (d >= 0)
        atomicMax(agg + (size_t)d * 128 + col, __float_as_uint(v));
    }
  }
}

// ---------------------------------------------------------------------------
// Node kernel: out = relu(agg @ gw1 + gb1).  8 row-tiles of 16 nodes / block.
// ---------------------------------------------------------------------------
__global__ __launch_bounds__(256)
void node_gemm(const float* __restrict__ agg, const _Float16* __restrict__ gw1p,
               const float* __restrict__ gb1, float* __restrict__ out) {
  __shared__ _Float16 sA[8 * 16 * 128];   // 32 KB: per-wave A tile (f16)

  const int tid  = threadIdx.x;
  const int wave = tid >> 5;
  const int lane = tid & 31;
  _Float16* a = sA + wave * (16 * 128);
  const int rt = blockIdx.x * 8 + wave;
  const bool valid = rt < (N_NODES / 16);   // 50000/16 = 3125 exact

  if (valid) {
    const int r  = lane & 15;
    const int kb = (lane >> 4) * 64;
    const float4* row = (const float4*)(agg + ((size_t)rt * 16 + r) * 128 + kb);
#pragma unroll
    for (int q = 0; q < 16; ++q) {
      float4 v = row[q];
      v4h h;
      h[0] = (_Float16)v.x; h[1] = (_Float16)v.y;
      h[2] = (_Float16)v.z; h[3] = (_Float16)v.w;
      *(v4h*)(a + r * 128 + kb + q * 4) = h;
    }
  }
  __syncthreads();
  if (!valid) return;

  v16h aF[4];
#pragma unroll
  for (int kt = 0; kt < 4; ++kt) aF[kt] = load_a_frag_lds(a, 128, kt * 32, lane);

  const int colBase = lane & 15;
  const int rowOff  = (lane >> 4) * 8;
#pragma unroll 4
  for (int nt = 0; nt < 16; ++nt) {
    v8f c = {};
#pragma unroll
    for (int kt = 0; kt < 4; ++kt)
      c = wmma_f16(aF[kt], load_b_frag(gw1p, kt * 16 + nt, lane), c);
    const int col = nt * 16 + colBase;
    const float bb = gb1[col];
#pragma unroll
    for (int r = 0; r < 8; ++r) {
      float v = c[r] + bb;
      v = v > 0.f ? v : 0.f;
      out[((size_t)rt * 16 + rowOff + r) * 256 + col] = v;
    }
  }
}

// ---------------------------------------------------------------------------
// Launch
// ---------------------------------------------------------------------------
extern "C" void kernel_launch(void* const* d_in, const int* in_sizes, int n_in,
                              void* d_out, int out_size, void* d_ws, size_t ws_size,
                              hipStream_t stream) {
  (void)in_sizes; (void)n_in; (void)out_size; (void)ws_size;
  const float*     x    = (const float*)d_in[0];
  const float*     pos  = (const float*)d_in[1];
  const long long* src  = (const long long*)d_in[2];
  const long long* dst  = (const long long*)d_in[3];
  const float*     lw1  = (const float*)d_in[4];
  const float*     lb1  = (const float*)d_in[5];
  const float*     lw2  = (const float*)d_in[6];
  const float*     lb2  = (const float*)d_in[7];
  const float*     gw1  = (const float*)d_in[8];
  const float*     gb1  = (const float*)d_in[9];
  float* out = (float*)d_out;

  char* ws = (char*)d_ws;
  float*     agg  = (float*)(ws + AGG_OFF);
  _Float16*  lw1p = (_Float16*)(ws + LW1P_OFF);
  _Float16*  lw2p = (_Float16*)(ws + LW2P_OFF);
  _Float16*  gw1p = (_Float16*)(ws + GW1P_OFF);

  // phase 0: init agg (relu => max >= 0, so 0-init is exact) + pack weights f16
  const int aggN = N_NODES * 128;
  zero_agg_kernel<<<(aggN + 255) / 256, 256, 0, stream>>>(agg, aggN);
  pack_w_kernel<<<(6144 + 255) / 256, 256, 0, stream>>>(lw1, lw1p, 67, 64, 4, 6144);
  pack_w_kernel<<<(8192 + 255) / 256, 256, 0, stream>>>(lw2, lw2p, 64, 128, 8, 8192);
  pack_w_kernel<<<(32768 + 255) / 256, 256, 0, stream>>>(gw1, gw1p, 128, 256, 16, 32768);

  // phase 1: edge MLP + scatter-max (128 edges per block)
  const int edgeBlocks = (EA + 127) / 128;     // 12891
  edge_mlp_scatter<<<edgeBlocks, 256, 0, stream>>>(
      x, pos, src, dst, lw1p, lb1, lw2p, lb2, (unsigned int*)agg);

  // phase 2: node GEMM + relu (3125 row tiles, 8 per block)
  node_gemm<<<(3125 + 7) / 8, 256, 0, stream>>>(agg, gw1p, gb1, out);
}
